// CausalFieldLayer_51410758533306
// MI455X (gfx1250) — matrix operand order repaired and checked by reading, not compiled, not run on hardware
//
#include <hip/hip_runtime.h>
#include <hip/hip_bf16.h>

// ---------------------------------------------------------------------------
// CausalFieldLayer for MI455X (gfx1250): bf16 WMMA GEMMs + fused epilogues.
// B=8, N=4096, D_MODEL=1024, D_FIELD=16, KSIZE=64.
// ---------------------------------------------------------------------------

typedef __attribute__((ext_vector_type(16))) __bf16 v16bf;
typedef __attribute__((ext_vector_type(8)))  __bf16 v8bf;
typedef __attribute__((ext_vector_type(8)))  float  v8f;

#define TOKENS   32768     // B*N
#define DMODEL   1024
#define DOUTK    256       // D_FIELD*D_FIELD
#define SEQN     4096
#define NBATCH   8

__device__ __forceinline__ unsigned short f32_bf16(float f) {
  unsigned u = __builtin_bit_cast(unsigned, f);
  u += 0x7FFFu + ((u >> 16) & 1u);            // round-to-nearest-even
  return (unsigned short)(u >> 16);
}
__device__ __forceinline__ float bf16_f32(unsigned short h) {
  unsigned u = ((unsigned)h) << 16;
  return __builtin_bit_cast(float, u);
}

__device__ __forceinline__ v8f wmma_bf16(v16bf a, v16bf b, v8f c) {
  // D = A(16x32 bf16) * B(32x16 bf16) + C(16x16 f32)
  return __builtin_amdgcn_wmma_f32_16x16x32_bf16(false, a, false, b,
                                                 (short)0, c, false, false);
}

union frag16 { v16bf v; v8bf half8[2]; };

// A fragment, 16x32 bf16, row-major source with row stride ld (elements).
// Layout (ISA 7.12.2): lanes 0-15 row=m, K {k0..k0+7, k0+16..k0+23};
//                      lanes 16-31 row=m, K {k0+8..15, k0+24..31}.
// Two 16B loads land directly in the fragment's register halves.
__device__ __forceinline__ v16bf load_a_bf16(const unsigned short* base, int row,
                                             int ld, int k0, int lane) {
  int m = lane & 15, sel = (lane >> 4) & 1;
  const unsigned short* p = base + (size_t)(row + m) * ld + k0 + sel * 8;
  frag16 f;
  f.half8[0] = *(const v8bf*)p;
  f.half8[1] = *(const v8bf*)(p + 16);
  return f.v;
}

// B fragment, 32x16 bf16, from pre-transposed weight Wt[n][k] (row stride ld).
// Lanes 0-15: col n0+l, K {k0..k0+7, k0+16..k0+23}; lanes 16-31: col n0+l-16, K {k0+8..15, k0+24..31}.
__device__ __forceinline__ v16bf load_b_t(const unsigned short* Wt, int n0,
                                          int ld, int k0, int lane) {
  int n = n0 + (lane & 15);
  int sel = (lane >> 4) & 1;
  const unsigned short* p = Wt + (size_t)n * ld + k0 + sel * 8;
  frag16 f;
  f.half8[0] = *(const v8bf*)p;
  f.half8[1] = *(const v8bf*)(p + 16);
  return f.v;
}

// ---------------------------------------------------------------------------
// K0a: weight transpose + f32->bf16 conversion (weights stay L2-resident).
// ---------------------------------------------------------------------------
__global__ __launch_bounds__(256) void k0_convert(
    const float* __restrict__ W_in, const float* __restrict__ W_sigma,
    const float* __restrict__ W_lam, const float* __restrict__ W_alp,
    const float* __restrict__ W_out,
    unsigned short* __restrict__ WtIn, unsigned short* __restrict__ WtCat,
    unsigned short* __restrict__ WtOut) {
  const int NI = DMODEL * DMODEL, NC = 48 * DMODEL, NO = DMODEL * DOUTK;
  int idx = blockIdx.x * 256 + threadIdx.x;
  if (idx < NI) {                               // WtIn[n][k] = W_in[k][n]
    int n = idx >> 10, k = idx & 1023;
    WtIn[idx] = f32_bf16(W_in[(size_t)k * DMODEL + n]);
  } else if (idx < NI + NC) {                   // WtCat[j][k], j = sigma|lam|alp
    int t = idx - NI; int j = t >> 10, k = t & 1023;
    const float* src = (j < 16) ? W_sigma : (j < 32) ? W_lam : W_alp;
    WtCat[t] = f32_bf16(src[(size_t)k * 16 + (j & 15)]);
  } else if (idx < NI + NC + NO) {              // WtOut[n][k] = W_out[k][n]
    int t = idx - NI - NC; int n = t >> 8, k = t & 255;
    WtOut[t] = f32_bf16(W_out[(size_t)k * DMODEL + n]);
  }
}

// ---------------------------------------------------------------------------
// K0b: x f32 -> bf16, vectorized (float4 in, ushort4 out). One-time pass so
// the 68.7 GF GEMM never pays conversion VALU and reads half the A bytes.
// ---------------------------------------------------------------------------
__global__ __launch_bounds__(256) void k0_convert_x(
    const float* __restrict__ x, unsigned short* __restrict__ xbf) {
  size_t i = (size_t)blockIdx.x * 256 + threadIdx.x;   // one float4 per thread
  float4 v = ((const float4*)x)[i];
  ushort4 o;
  o.x = f32_bf16(v.x); o.y = f32_bf16(v.y);
  o.z = f32_bf16(v.z); o.w = f32_bf16(v.w);
  ((ushort4*)xbf)[i] = o;
}

// ---------------------------------------------------------------------------
// K1: h = x @ W_in + b_in   (32768 x 1024 x 1024), all operands bf16 in loop.
// Block = 8 waves, tile 128x128, wave sub-tile 32x64 (2x4 WMMA frags).
// ---------------------------------------------------------------------------
__global__ __launch_bounds__(256) void k1_gemm_h(
    const unsigned short* __restrict__ xbf, const unsigned short* __restrict__ WtIn,
    const float* __restrict__ b_in, unsigned short* __restrict__ h) {
  const int lane = threadIdx.x & 31;
  const int w    = threadIdx.x >> 5;
  const int m0 = blockIdx.y * 128 + (w & 3) * 32;
  const int n0 = blockIdx.x * 128 + (w >> 2) * 64;

  v8f acc[2][4];
  #pragma unroll
  for (int mi = 0; mi < 2; ++mi)
    #pragma unroll
    for (int nj = 0; nj < 4; ++nj)
      #pragma unroll
      for (int e = 0; e < 8; ++e) acc[mi][nj][e] = 0.f;

  for (int k0 = 0; k0 < DMODEL; k0 += 32) {
    v16bf a0 = load_a_bf16(xbf, m0,      DMODEL, k0, lane);
    v16bf a1 = load_a_bf16(xbf, m0 + 16, DMODEL, k0, lane);
    #pragma unroll
    for (int nj = 0; nj < 4; ++nj) {
      v16bf b = load_b_t(WtIn, n0 + 16 * nj, DMODEL, k0, lane);
      acc[0][nj] = wmma_bf16(a0, b, acc[0][nj]);
      acc[1][nj] = wmma_bf16(a1, b, acc[1][nj]);
    }
  }
  const int cb = lane & 15, rsel = (lane >> 4) * 8;
  #pragma unroll
  for (int nj = 0; nj < 4; ++nj) {
    int col = n0 + 16 * nj + cb;
    float bb = b_in[col];
    #pragma unroll
    for (int mi = 0; mi < 2; ++mi)
      #pragma unroll
      for (int i = 0; i < 8; ++i) {
        int row = m0 + mi * 16 + rsel + i;
        h[(size_t)row * DMODEL + col] = f32_bf16(acc[mi][nj][i] + bb);
      }
  }
}

// ---------------------------------------------------------------------------
// K2: P = h @ [W_sigma|W_lam|W_alp] + biases   (32768 x 1024 x 48), f32 out.
// ---------------------------------------------------------------------------
__global__ __launch_bounds__(256) void k2_proj(
    const unsigned short* __restrict__ h, const unsigned short* __restrict__ WtCat,
    const float* __restrict__ b_sigma, const float* __restrict__ b_lam,
    const float* __restrict__ b_alp, float* __restrict__ P) {
  const int lane = threadIdx.x & 31;
  const int w    = threadIdx.x >> 5;
  const int r0   = blockIdx.x * 128 + w * 16;

  v8f acc[3];
  #pragma unroll
  for (int j = 0; j < 3; ++j)
    #pragma unroll
    for (int e = 0; e < 8; ++e) acc[j][e] = 0.f;

  for (int k0 = 0; k0 < DMODEL; k0 += 32) {
    v16bf a = load_a_bf16(h, r0, DMODEL, k0, lane);
    #pragma unroll
    for (int j = 0; j < 3; ++j)
      acc[j] = wmma_bf16(a, load_b_t(WtCat, 16 * j, DMODEL, k0, lane), acc[j]);
  }
  const int cb = lane & 15, rsel = (lane >> 4) * 8;
  #pragma unroll
  for (int j = 0; j < 3; ++j) {
    float bb = (j == 0) ? b_sigma[cb] : (j == 1) ? b_lam[cb] : b_alp[cb];
    int col = 16 * j + cb;
    #pragma unroll
    for (int i = 0; i < 8; ++i)
      P[(size_t)(r0 + rsel + i) * 48 + col] = acc[j][i] + bb;
  }
}

// ---------------------------------------------------------------------------
// K3: per-token octonion algebra -> Jv; J expand + antisym; Pi contractions.
// Writes Jflat (B,N,256) and tf (B,N,256) in f32. One thread per token.
// Structure constants hardcoded (7 Fano triples).
// ---------------------------------------------------------------------------
__device__ __forceinline__ void omul8(const float* a, const float* b, float* o) {
  #pragma unroll
  for (int i = 0; i < 8; ++i) o[i] = 0.f;
  const int T7[7][3] = {{0,1,2},{0,3,4},{0,5,6},{1,3,5},{1,4,6},{2,3,6},{2,4,5}};
  #pragma unroll
  for (int t = 0; t < 7; ++t) {
    int i = T7[t][0], j = T7[t][1], k = T7[t][2];
    o[k] += a[i] * b[j] - a[j] * b[i];
    o[i] += a[j] * b[k] - a[k] * b[j];
    o[j] += a[k] * b[i] - a[i] * b[k];
  }
}
__device__ __forceinline__ void cmul16(const float* u, const float* v, float* o) {
  float t1[8], t2[8];
  omul8(u, v, t1); omul8(u + 8, v + 8, t2);
  #pragma unroll
  for (int i = 0; i < 8; ++i) o[i] = t1[i] - t2[i];
  omul8(u, v + 8, t1); omul8(u + 8, v, t2);
  #pragma unroll
  for (int i = 0; i < 8; ++i) o[8 + i] = t1[i] + t2[i];
}

__global__ __launch_bounds__(256) void k3_token(
    const float* __restrict__ P, const float* __restrict__ J_expand,
    const float* __restrict__ tetrad, const float* __restrict__ gammas,
    const float* __restrict__ Pi_source, const float* __restrict__ Pi_target,
    const float* __restrict__ Pi_spinor,
    float* __restrict__ Jflat, float* __restrict__ tf) {
  const int r = blockIdx.x * 256 + threadIdx.x;
  float ps[16], pl[16], pa[16];
  #pragma unroll
  for (int i = 0; i < 16; ++i) {
    ps[i] = P[(size_t)r * 48 + i];
    pl[i] = P[(size_t)r * 48 + 16 + i];
    pa[i] = P[(size_t)r * 48 + 32 + i];
  }
  float u1[16], u2[16], u3[16], u4[16], Jv[16];
  cmul16(ps, pl, u1); cmul16(u1, pa, u2);          // (ps*pl)*pa
  cmul16(pl, pa, u3); cmul16(ps, u3, u4);          // ps*(pl*pa)
  #pragma unroll
  for (int i = 0; i < 16; ++i) Jv[i] = u2[i] - u4[i];

  float inter1[16];
  #pragma unroll
  for (int m = 0; m < 16; ++m) inter1[m] = 0.f;

  #pragma unroll 1
  for (int j = 0; j < 16; ++j)
    #pragma unroll 1
    for (int k = 0; k < 16; ++k) {
      float s = 0.f;
      #pragma unroll
      for (int i = 0; i < 16; ++i)
        s += Jv[i] * (J_expand[((size_t)i * 16 + j) * 16 + k] -
                      J_expand[((size_t)i * 16 + k) * 16 + j]);
      Jflat[(size_t)r * 256 + j * 16 + k] = s;
      #pragma unroll
      for (int m = 0; m < 16; ++m)
        inter1[m] += s * Pi_source[((size_t)j * 16 + k) * 16 + m];
    }

  // Gamma[c][d] = sum_{a,b} tetrad[a][b] * gammas[b][c][d];  sp[k] = <Pi_spinor, Gamma>
  float G[16];
  #pragma unroll
  for (int c = 0; c < 4; ++c)
    #pragma unroll
    for (int d = 0; d < 4; ++d) {
      float g = 0.f;
      #pragma unroll
      for (int a = 0; a < 4; ++a)
        #pragma unroll
        for (int b = 0; b < 4; ++b)
          g += tetrad[a * 4 + b] * gammas[(b * 4 + c) * 4 + d];
      G[c * 4 + d] = g;
    }
  float sp[16];
  #pragma unroll
  for (int k = 0; k < 16; ++k) {
    float s = 0.f;
    #pragma unroll
    for (int g = 0; g < 4; ++g)
      #pragma unroll
      for (int d = 0; d < 4; ++d)
        s += Pi_spinor[((size_t)g * 4 + d) * 16 + k] * G[g * 4 + d];
    sp[k] = s;
  }

  #pragma unroll 1
  for (int p = 0; p < 16; ++p)
    #pragma unroll 1
    for (int q = 0; q < 16; ++q) {
      float v = 0.f;
      #pragma unroll
      for (int k = 0; k < 16; ++k)
        v += inter1[k] * Pi_target[((size_t)p * 16 + q) * 16 + k];
      tf[(size_t)r * 256 + p * 16 + q] = v * sp[q];
    }
}

// ---------------------------------------------------------------------------
// K4: causal 64-tap FIR along n (== the rfft/irfft path, no circular wrap),
// then T = alpha*Jflat + (1-alpha)*mem, stored bf16. Sliding window in LDS.
// ---------------------------------------------------------------------------
__global__ __launch_bounds__(256) void k4_fir(
    const float* __restrict__ tf, const float* __restrict__ Jflat,
    const float* __restrict__ kw, unsigned short* __restrict__ T) {
  __shared__ unsigned short sw[127 * 256];      // 63.5 KB window, bf16
  const int c  = threadIdx.x;                   // channel
  const int b  = blockIdx.y;
  const int n0 = blockIdx.x * 64;
  const float* tfb = tf + (size_t)b * SEQN * 256;

  for (int ri = 0; ri < 127; ++ri) {
    int g = n0 - 63 + ri;
    float v = (g >= 0) ? tfb[(size_t)g * 256 + c] : 0.f;
    sw[ri * 256 + c] = f32_bf16(v);
  }
  __syncthreads();

  const float alpha = kw[0];
  for (int j = 0; j < 64; ++j) {
    float mem = 0.f;
    #pragma unroll 8
    for (int t = 0; t < 64; ++t)
      mem += kw[t] * bf16_f32(sw[(63 + j - t) * 256 + c]);
    size_t idx = ((size_t)b * SEQN + n0 + j) * 256 + c;
    T[idx] = f32_bf16(alpha * Jflat[idx] + (1.f - alpha) * mem);
  }
}

// ---------------------------------------------------------------------------
// K5: y = LN( x + T @ W_out + b_out ) * ln_g + ln_b.
// Block = 8 waves, 16 rows x 1024 cols (each wave 16x128, K=256).
// LayerNorm fused: half-wave shfl_xor row reduce + small LDS cross-wave.
// ---------------------------------------------------------------------------
__global__ __launch_bounds__(256) void k5_out_ln(
    const unsigned short* __restrict__ T, const unsigned short* __restrict__ WtOut,
    const float* __restrict__ b_out, const float* __restrict__ x,
    const float* __restrict__ ln_g, const float* __restrict__ ln_b,
    float* __restrict__ y) {
  __shared__ float part[2][8][16];
  const int lane = threadIdx.x & 31;
  const int w    = threadIdx.x >> 5;
  const int r0   = blockIdx.x * 16;
  const int n0   = w * 128;

  v8f acc[8];
  #pragma unroll
  for (int j = 0; j < 8; ++j)
    #pragma unroll
    for (int e = 0; e < 8; ++e) acc[j][e] = 0.f;

  for (int k0 = 0; k0 < DOUTK; k0 += 32) {
    v16bf a = load_a_bf16(T, r0, DOUTK, k0, lane);
    #pragma unroll
    for (int j = 0; j < 8; ++j)
      acc[j] = wmma_bf16(a, load_b_t(WtOut, n0 + 16 * j, DOUTK, k0, lane), acc[j]);
  }

  const int cb = lane & 15, rsel = (lane >> 4) * 8;
  float s[8], q[8];
  #pragma unroll
  for (int i = 0; i < 8; ++i) { s[i] = 0.f; q[i] = 0.f; }

  #pragma unroll
  for (int j = 0; j < 8; ++j) {
    int col = n0 + 16 * j + cb;
    float bb = b_out[col];
    #pragma unroll
    for (int i = 0; i < 8; ++i) {
      int row = r0 + rsel + i;
      float v = acc[j][i] + bb + x[(size_t)row * DMODEL + col];
      acc[j][i] = v;                          // keep y pre-norm in registers
      s[i] += v; q[i] += v * v;
    }
  }
  // reduce across the 16 lanes that share the same rows (masks < 16)
  #pragma unroll
  for (int m = 8; m >= 1; m >>= 1)
    #pragma unroll
    for (int i = 0; i < 8; ++i) {
      s[i] += __shfl_xor(s[i], m, 32);
      q[i] += __shfl_xor(q[i], m, 32);
    }
  if (cb == 0) {
    #pragma unroll
    for (int i = 0; i < 8; ++i) {
      part[0][w][rsel + i] = s[i];
      part[1][w][rsel + i] = q[i];
    }
  }
  __syncthreads();
  #pragma unroll
  for (int i = 0; i < 8; ++i) {
    float ts = 0.f, tq = 0.f;
    #pragma unroll
    for (int ww = 0; ww < 8; ++ww) {
      ts += part[0][ww][rsel + i];
      tq += part[1][ww][rsel + i];
    }
    float mu   = ts * (1.f / 1024.f);
    float var  = tq * (1.f / 1024.f) - mu * mu;
    float rstd = rsqrtf(var + 1e-5f);
    int row = r0 + rsel + i;
    #pragma unroll
    for (int j = 0; j < 8; ++j) {
      int col = n0 + 16 * j + cb;
      y[(size_t)row * DMODEL + col] = (acc[j][i] - mu) * rstd * ln_g[col] + ln_b[col];
    }
  }
}

// ---------------------------------------------------------------------------
extern "C" void kernel_launch(void* const* d_in, const int* in_sizes, int n_in,
                              void* d_out, int out_size, void* d_ws, size_t ws_size,
                              hipStream_t stream) {
  (void)in_sizes; (void)n_in; (void)out_size; (void)ws_size;
  const float* x        = (const float*)d_in[0];
  const float* W_in     = (const float*)d_in[1];
  const float* b_in     = (const float*)d_in[2];
  const float* W_sigma  = (const float*)d_in[3];
  const float* b_sigma  = (const float*)d_in[4];
  const float* W_lam    = (const float*)d_in[5];
  const float* b_lam    = (const float*)d_in[6];
  const float* W_alp    = (const float*)d_in[7];
  const float* b_alp    = (const float*)d_in[8];
  /* d_in[9] = oct_struct: structure constants hardcoded in k3 */
  const float* J_expand  = (const float*)d_in[10];
  const float* tetrad    = (const float*)d_in[11];
  const float* gammas    = (const float*)d_in[12];
  const float* Pi_source = (const float*)d_in[13];
  const float* Pi_target = (const float*)d_in[14];
  const float* Pi_spinor = (const float*)d_in[15];
  const float* kweights  = (const float*)d_in[16];
  const float* W_out     = (const float*)d_in[17];
  const float* b_out     = (const float*)d_in[18];
  const float* ln_g      = (const float*)d_in[19];
  const float* ln_b      = (const float*)d_in[20];
  float* y = (float*)d_out;

  // workspace layout (bytes), all 256B-aligned
  char* ws = (char*)d_ws;
  unsigned short* WtIn  = (unsigned short*)(ws + 0);                 // 2 MB
  unsigned short* WtCat = (unsigned short*)(ws + 2097152);           // 96 KB
  unsigned short* WtOut = (unsigned short*)(ws + 2195456);           // 512 KB
  unsigned short* xbf   = (unsigned short*)(ws + 2719744);           // 64 MB
  unsigned short* h     = (unsigned short*)(ws + 69828608);          // 64 MB
  float*          Pp    = (float*)(ws + 136937472);                  // 6 MB
  float*          Jfl   = (float*)(ws + 143228928);                  // 32 MB
  float*          tfv   = (float*)(ws + 176783360);                  // 32 MB
  unsigned short* Tt    = (unsigned short*)(ws + 210337792);         // 16 MB

  const int conv_total = DMODEL * DMODEL + 48 * DMODEL + DMODEL * DOUTK;
  k0_convert<<<(conv_total + 255) / 256, 256, 0, stream>>>(
      W_in, W_sigma, W_lam, W_alp, W_out, WtIn, WtCat, WtOut);

  // 32768*1024 elements, 4 per thread
  k0_convert_x<<<(TOKENS * DMODEL / 4) / 256, 256, 0, stream>>>(x, xbf);

  k1_gemm_h<<<dim3(DMODEL / 128, TOKENS / 128), 256, 0, stream>>>(xbf, WtIn, b_in, h);

  k2_proj<<<TOKENS / 128, 256, 0, stream>>>(h, WtCat, b_sigma, b_lam, b_alp, Pp);

  k3_token<<<TOKENS / 256, 256, 0, stream>>>(Pp, J_expand, tetrad, gammas,
                                             Pi_source, Pi_target, Pi_spinor,
                                             Jfl, tfv);

  k4_fir<<<dim3(SEQN / 64, NBATCH), 256, 0, stream>>>(tfv, Jfl, kweights, Tt);

  k5_out_ln<<<TOKENS / 16, 256, 0, stream>>>(Tt, WtOut, b_out, x, ln_g, ln_b, y);
}
